// DeformAttnwMotion_68332929679389
// MI455X (gfx1250) — matrix operand order repaired and checked by reading, not compiled, not run on hardware
//
#include <hip/hip_runtime.h>

// ---------------------------------------------------------------------------
// DeformAttnwMotion for MI455X (gfx1250, wave32, WMMA).
//
//   Kernel 1 : Q projection   — bf16 split-precision WMMA GEMM
//   Kernel 2 : K+V projection — fused: kv tensor read ONCE, X tile staged to
//              LDS via global_load_async_to_lds_b128 (ASYNCcnt), both weight
//              matrices split hi/lo in LDS, each A fragment feeds 6 WMMAs.
//   Kernel 3 : per-pixel 8-head / 18-sample attention + softmax outputs.
//
// HBM-bound problem (~0.8 GB traffic vs 20 GFLOP after K/V fusion), so the
// precision path is chosen for accuracy: 3-pass bf16 splitting
// (Ah*Bh + Ah*Bl + Al*Bh) recovers ~fp32 accuracy at negligible extra matrix
// work. A/B fragments use contiguous k = 16*(lane/16) + e, same in both.
// ---------------------------------------------------------------------------

#define NSAMP 18
#define NHEAD 8
#define HEADC 16
#define OUTC  128
#define CIN   128
#define DPIX  4096
#define NB    4

typedef __attribute__((ext_vector_type(16))) __bf16 v16bf;
typedef __attribute__((ext_vector_type(8)))  float  v8f;

// Issue one async 16-byte global->LDS copy (tracked by ASYNCcnt).
static __device__ __forceinline__ void async_cp_b128(unsigned lds_byte_off,
                                                     const float* gaddr) {
  unsigned long long ga = (unsigned long long)(size_t)gaddr;
  asm volatile("global_load_async_to_lds_b128 %0, %1, off"
               :: "v"(lds_byte_off), "v"(ga) : "memory");
}
static __device__ __forceinline__ void wait_async0() {
  asm volatile("s_wait_asynccnt 0x0" ::: "memory");
}

// Stage a 128x128 fp32 X tile [c][m] into LDS via async DMA (16 B128/thread).
static __device__ __forceinline__ void stage_x_async(float* sX, const float* Xb,
                                                     int R, int tid) {
  const unsigned sXbase = (unsigned)(size_t)(void*)sX;  // low 32 bits == LDS off
#pragma unroll
  for (int i = 0; i < 16; ++i) {
    int q  = tid + i * 256;   // 16B chunk id, 0..4095
    int c  = q >> 5;
    int mc = q & 31;
    async_cp_b128(sXbase + (unsigned)q * 16u, Xb + (size_t)c * R + mc * 4);
  }
}

// Split one 128x128 weight matrix into bf16 hi/lo in LDS (64 elems/thread).
static __device__ __forceinline__ void stage_w_split(__bf16* sWh, __bf16* sWl,
                                                     const float* W, int tid) {
#pragma unroll 8
  for (int i = 0; i < 64; ++i) {
    int idx = tid + i * 256;          // 0..16383 == oc*128 + c
    float wv = W[idx];
    __bf16 h = (__bf16)wv;            // native RTNE cvt
    sWh[idx] = h;
    sWl[idx] = (__bf16)(wv - (float)h);
  }
}

// Build the A fragment pair (hi/lo) for one 16x32 k-tile from the fp32 LDS
// tile. Element e -> K = kt*32 + (hi?16:0) + e (matches B fragment).
static __device__ __forceinline__ void build_a_frag(v16bf& Ah, v16bf& Al,
                                                    const float* sX, int kt,
                                                    int mrow, int hihalf) {
#pragma unroll
  for (int e = 0; e < 16; ++e) {
    int k = kt * 32 + (hihalf ? 16 : 0) + e;
    float v  = sX[k * 128 + mrow];
    __bf16 h = (__bf16)v;
    Ah[e] = h;
    Al[e] = (__bf16)(v - (float)h);
  }
}

// Load the B fragment pair for one (kt, nt) from split-W LDS.
static __device__ __forceinline__ void load_b_frag(v16bf& Bh, v16bf& Bl,
                                                   const __bf16* sWh,
                                                   const __bf16* sWl,
                                                   int off) {
#pragma unroll
  for (int e = 0; e < 16; ++e) {
    Bh[e] = sWh[off + e];
    Bl[e] = sWl[off + e];
  }
}

// 3-pass split-precision accumulate: D += Ah*Bh + Ah*Bl + Al*Bh
static __device__ __forceinline__ v8f wmma3(v16bf Ah, v16bf Al, v16bf Bh,
                                            v16bf Bl, v8f acc) {
  acc = __builtin_amdgcn_wmma_f32_16x16x32_bf16(false, Ah, false, Bh, (short)0,
                                                acc, false, false);
  acc = __builtin_amdgcn_wmma_f32_16x16x32_bf16(false, Ah, false, Bl, (short)0,
                                                acc, false, false);
  acc = __builtin_amdgcn_wmma_f32_16x16x32_bf16(false, Al, false, Bh, (short)0,
                                                acc, false, false);
  return acc;
}

// ---------------------------------------------------------------------------
// Single-matrix projection (used for Q):
//   X (NB, CIN, R) ; W (OUTC, CIN) ; Y (NB, R, OUTC)
// ---------------------------------------------------------------------------
__global__ __launch_bounds__(256) void proj_gemm_kernel(
    const float* __restrict__ X, const float* __restrict__ W,
    const float* __restrict__ bias, float* __restrict__ Y, int R) {
  __shared__ float  sX[CIN * 128];
  __shared__ __bf16 sWh[OUTC * CIN];
  __shared__ __bf16 sWl[OUTC * CIN];

  const int tid  = threadIdx.x;
  const int lane = tid & 31;
  const int wave = tid >> 5;
  const int b    = blockIdx.y;
  const int r0   = blockIdx.x * 128;

  const float* Xb = X + (size_t)b * CIN * (size_t)R + r0;
  stage_x_async(sX, Xb, R, tid);
  __builtin_prefetch((const char*)W + (size_t)tid * 256, 0, 3);
  stage_w_split(sWh, sWl, W, tid);
  wait_async0();
  __syncthreads();

  v8f acc[8];
#pragma unroll
  for (int nt = 0; nt < 8; ++nt)
#pragma unroll
    for (int j = 0; j < 8; ++j) acc[nt][j] = 0.0f;

  const int mrow   = wave * 16 + (lane & 15);
  const int hihalf = (lane < 16) ? 0 : 1;

#pragma unroll
  for (int kt = 0; kt < 4; ++kt) {
    v16bf Ah, Al;
    build_a_frag(Ah, Al, sX, kt, mrow, hihalf);
    const int kbase = kt * 32 + (hihalf ? 16 : 0);
#pragma unroll
    for (int nt = 0; nt < 8; ++nt) {
      const int off = (nt * 16 + (lane & 15)) * CIN + kbase;
      v16bf Bh, Bl;
      load_b_frag(Bh, Bl, sWh, sWl, off);
      acc[nt] = wmma3(Ah, Al, Bh, Bl, acc[nt]);
    }
  }

#pragma unroll
  for (int nt = 0; nt < 8; ++nt) {
    const int n    = nt * 16 + (lane & 15);
    const float bn = bias[n];
#pragma unroll
    for (int j = 0; j < 8; ++j) {
      int m = wave * 16 + j + (hihalf ? 8 : 0);
      size_t row = (size_t)b * R + (size_t)(r0 + m);
      Y[row * OUTC + n] = acc[nt][j] + bn;
    }
  }
}

// ---------------------------------------------------------------------------
// Fused K+V projection: kv tensor read once; each A fragment feeds 6 WMMAs.
// LDS: 64 KB X tile + 2 x 64 KB split weights = 192 KB (within 320 KB/WGP).
// ---------------------------------------------------------------------------
__global__ __launch_bounds__(256) void proj_kv_kernel(
    const float* __restrict__ X,
    const float* __restrict__ Wk, const float* __restrict__ bk,
    const float* __restrict__ Wv, const float* __restrict__ bv,
    float* __restrict__ Yk, float* __restrict__ Yv, int R) {
  __shared__ float  sX[CIN * 128];     // 64 KB
  __shared__ __bf16 sKh[OUTC * CIN];   // 32 KB
  __shared__ __bf16 sKl[OUTC * CIN];   // 32 KB
  __shared__ __bf16 sVh[OUTC * CIN];   // 32 KB
  __shared__ __bf16 sVl[OUTC * CIN];   // 32 KB

  const int tid  = threadIdx.x;
  const int lane = tid & 31;
  const int wave = tid >> 5;
  const int b    = blockIdx.y;
  const int r0   = blockIdx.x * 128;

  const float* Xb = X + (size_t)b * CIN * (size_t)R + r0;
  stage_x_async(sX, Xb, R, tid);
  __builtin_prefetch((const char*)Wk + (size_t)tid * 256, 0, 3);
  __builtin_prefetch((const char*)Wv + (size_t)tid * 256, 0, 3);
  stage_w_split(sKh, sKl, Wk, tid);
  stage_w_split(sVh, sVl, Wv, tid);
  wait_async0();
  __syncthreads();

  v8f accK[8], accV[8];
#pragma unroll
  for (int nt = 0; nt < 8; ++nt)
#pragma unroll
    for (int j = 0; j < 8; ++j) { accK[nt][j] = 0.0f; accV[nt][j] = 0.0f; }

  const int mrow   = wave * 16 + (lane & 15);
  const int hihalf = (lane < 16) ? 0 : 1;

#pragma unroll
  for (int kt = 0; kt < 4; ++kt) {
    v16bf Ah, Al;
    build_a_frag(Ah, Al, sX, kt, mrow, hihalf);
    const int kbase = kt * 32 + (hihalf ? 16 : 0);
#pragma unroll
    for (int nt = 0; nt < 8; ++nt) {
      const int off = (nt * 16 + (lane & 15)) * CIN + kbase;
      v16bf Bh, Bl;
      load_b_frag(Bh, Bl, sKh, sKl, off);
      accK[nt] = wmma3(Ah, Al, Bh, Bl, accK[nt]);
      load_b_frag(Bh, Bl, sVh, sVl, off);
      accV[nt] = wmma3(Ah, Al, Bh, Bl, accV[nt]);
    }
  }

#pragma unroll
  for (int nt = 0; nt < 8; ++nt) {
    const int n    = nt * 16 + (lane & 15);
    const float bkn = bk[n];
    const float bvn = bv[n];
#pragma unroll
    for (int j = 0; j < 8; ++j) {
      int m = wave * 16 + j + (hihalf ? 8 : 0);
      size_t row = (size_t)b * R + (size_t)(r0 + m);
      Yk[row * OUTC + n] = accK[nt][j] + bkn;
      Yv[row * OUTC + n] = accV[nt][j] + bvn;
    }
  }
}

// ---------------------------------------------------------------------------
// Attention: one thread per (b, head, pixel); h fastest so 8 adjacent threads
// cover one pixel's full 512 B projected row -> coalesced b128 loads.
// ---------------------------------------------------------------------------
static __device__ __forceinline__ float dot4(float4 a, float4 b) {
  return a.x * b.x + a.y * b.y + a.z * b.z + a.w * b.w;
}
static __device__ __forceinline__ void fma4(float4& a, float s, float4 v) {
  a.x += s * v.x; a.y += s * v.y; a.z += s * v.z; a.w += s * v.w;
}

__global__ __launch_bounds__(256) void attn_kernel(
    const float* __restrict__ qp,   // (NB, DPIX, OUTC)
    const float* __restrict__ kp,   // (NB, NSAMP*DPIX, OUTC)
    const float* __restrict__ vp,   // (NB, NSAMP*DPIX, OUTC)
    float* __restrict__ out,        // (NB, OUTC, DPIX)
    float* __restrict__ kv0,        // (NB*NHEAD, 9, DPIX)
    float* __restrict__ kv1) {      // (NB*NHEAD, 9, DPIX)
  const int g = blockIdx.x * blockDim.x + threadIdx.x;
  const int h = g & 7;
  const int f = (g >> 3) & (DPIX - 1);
  const int b = g >> 15;                                  // DPIX*NHEAD = 2^15

  const float scale = 0.25f;                              // HEADC^-0.5

  const float4* q4 = (const float4*)(qp + ((size_t)b * DPIX + f) * OUTC + h * HEADC);
  float4 q0 = q4[0], q1 = q4[1], q2 = q4[2], q3 = q4[3];

  float attn[NSAMP];
#pragma unroll
  for (int s = 0; s < NSAMP; ++s) {
    const float4* k4 = (const float4*)(kp + (((size_t)b * NSAMP + s) * DPIX + f) * OUTC + h * HEADC);
    float a = dot4(q0, k4[0]) + dot4(q1, k4[1]) + dot4(q2, k4[2]) + dot4(q3, k4[3]);
    attn[s] = a * scale;
  }

  float mx = attn[0];
#pragma unroll
  for (int s = 1; s < NSAMP; ++s) mx = fmaxf(mx, attn[s]);
  float e[NSAMP], sum = 0.0f;
#pragma unroll
  for (int s = 0; s < NSAMP; ++s) { e[s] = __expf(attn[s] - mx); sum += e[s]; }
  const float inv = 1.0f / sum;

  float4 a0 = {0, 0, 0, 0}, a1 = {0, 0, 0, 0}, a2 = {0, 0, 0, 0}, a3 = {0, 0, 0, 0};
#pragma unroll
  for (int s = 0; s < NSAMP; ++s) {
    const float sc = e[s] * inv;
    const float4* v4 = (const float4*)(vp + (((size_t)b * NSAMP + s) * DPIX + f) * OUTC + h * HEADC);
    fma4(a0, sc, v4[0]); fma4(a1, sc, v4[1]); fma4(a2, sc, v4[2]); fma4(a3, sc, v4[3]);
  }

  float* orow = out + ((size_t)b * OUTC + h * HEADC) * DPIX + f;
  orow[0 * DPIX]  = a0.x; orow[1 * DPIX]  = a0.y; orow[2 * DPIX]  = a0.z; orow[3 * DPIX]  = a0.w;
  orow[4 * DPIX]  = a1.x; orow[5 * DPIX]  = a1.y; orow[6 * DPIX]  = a1.z; orow[7 * DPIX]  = a1.w;
  orow[8 * DPIX]  = a2.x; orow[9 * DPIX]  = a2.y; orow[10 * DPIX] = a2.z; orow[11 * DPIX] = a2.w;
  orow[12 * DPIX] = a3.x; orow[13 * DPIX] = a3.y; orow[14 * DPIX] = a3.z; orow[15 * DPIX] = a3.w;

  // Half softmaxes: the global-max shift cancels, so reuse e[].
  float s0 = 0.0f, s1 = 0.0f;
#pragma unroll
  for (int s = 0; s < 9; ++s)  s0 += e[s];
#pragma unroll
  for (int s = 9; s < NSAMP; ++s) s1 += e[s];
  const float i0 = 1.0f / s0, i1 = 1.0f / s1;
  float* r0 = kv0 + ((size_t)(b * NHEAD + h) * 9) * DPIX + f;
  float* r1 = kv1 + ((size_t)(b * NHEAD + h) * 9) * DPIX + f;
#pragma unroll
  for (int s = 0; s < 9; ++s) r0[(size_t)s * DPIX] = e[s] * i0;
#pragma unroll
  for (int s = 0; s < 9; ++s) r1[(size_t)s * DPIX] = e[s + 9] * i1;
}

// ---------------------------------------------------------------------------
extern "C" void kernel_launch(void* const* d_in, const int* in_sizes, int n_in,
                              void* d_out, int out_size, void* d_ws, size_t ws_size,
                              hipStream_t stream) {
  (void)in_sizes; (void)n_in; (void)out_size; (void)ws_size;

  const float* q  = (const float*)d_in[0];  // (4,128,64,64)
  const float* kv = (const float*)d_in[1];  // (4,128,18,4096)
  const float* Wq = (const float*)d_in[2];
  const float* bq = (const float*)d_in[3];
  const float* Wk = (const float*)d_in[4];
  const float* bk = (const float*)d_in[5];
  const float* Wv = (const float*)d_in[6];
  const float* bv = (const float*)d_in[7];

  float* ws = (float*)d_ws;
  const size_t qpN = (size_t)NB * DPIX * OUTC;           // 2,097,152 floats
  const size_t kpN = (size_t)NB * NSAMP * DPIX * OUTC;   // 37,748,736 floats
  float* qp = ws;
  float* kp = ws + qpN;
  float* vp = kp + kpN;

  float* outp = (float*)d_out;
  float* o_main = outp;                                   // (4,128,64,64)
  float* o_kv0  = outp + (size_t)NB * OUTC * DPIX;        // (32,9,4096)
  float* o_kv1  = o_kv0 + (size_t)NB * NHEAD * 9 * DPIX;  // (32,9,4096)

  dim3 blk(256);
  proj_gemm_kernel<<<dim3(DPIX / 128, NB), blk, 0, stream>>>(q, Wq, bq, qp, DPIX);
  proj_kv_kernel<<<dim3(NSAMP * DPIX / 128, NB), blk, 0, stream>>>(
      kv, Wk, bk, Wv, bv, kp, vp, NSAMP * DPIX);
  attn_kernel<<<dim3((NB * DPIX * NHEAD) / 256), blk, 0, stream>>>(qp, kp, vp, o_main, o_kv0, o_kv1);
}
